// EncoderImageSg_66030827208767
// MI455X (gfx1250) — compile-verified, matrix operand order checked by reading, not run on hardware
//
#include <hip/hip_runtime.h>
#include <stdint.h>
#include <stddef.h>

// ---------------- problem constants (from reference) ----------------
#define O_N   2000
#define T_N   6000
#define DIN   2048
#define DW    300
#define DG    512
#define HH    512
#define L_EXTRA 4
#define KF    (DIN + DW)    // 2348
#define KFP   2368          // KF padded to multiple of 32
#define OPAD  2048          // 2000 -> 16 * 128
#define TPAD  6016          // 6000 -> 47 * 128

// ---------------- types ----------------
typedef __bf16   bf16_t;
typedef bf16_t   v16bf __attribute__((ext_vector_type(16)));
typedef float    v8f   __attribute__((ext_vector_type(8)));
typedef uint32_t v8u   __attribute__((ext_vector_type(8)));

__device__ __forceinline__ unsigned short f2bf(float f) {
  uint32_t u = __float_as_uint(f);
  u += 0x7FFFu + ((u >> 16) & 1u);          // round-to-nearest-even
  return (unsigned short)(u >> 16);
}

// ---------------- WMMA GEMM: C = act(A_bf16 @ Bt_bf16^T + bias) -------------
// A : Mpad x K bf16 row-major (lda = K stride)
// Bt: N x K bf16 row-major   (ldb = K stride)  == W pre-transposed
// Block tile 128x128, BK=32. 8 waves; wave -> 32x64 subtile = 2x4 WMMA frags.
// Double-buffered LDS fed by GLOBAL_LOAD_ASYNC_TO_LDS_B128 (ASYNCcnt pipeline),
// unrolled x2 so buffer indices are compile-time. Requires K/32 even (all our
// K: 2368/32=74, 6144/32=192, 512/32=16, 1536/32=48).
#define BM 128
#define BN 128
#define BK 32
#define LDS_STR (BK + 8)   // 40 halves -> 80B rows (16B aligned chunks)

template<bool RELU, bool OUT_BF16>
__global__ __launch_bounds__(256)
void gemm_bf16_wmma(const unsigned short* __restrict__ A, int lda,
                    const unsigned short* __restrict__ Bt, int ldb,
                    const float* __restrict__ bias,
                    void* __restrict__ Cout, int ldc, int Mstore, int K)
{
  __shared__ __align__(16) unsigned short As[2][BM * LDS_STR];
  __shared__ __align__(16) unsigned short Bs[2][BN * LDS_STR];

  const int tid  = threadIdx.x;
  const int lane = tid & 31;
  const int wave = tid >> 5;          // 0..7
  const int l    = lane & 15;
  const int half = lane >> 4;
  const int wm   = wave >> 1;         // 0..3 -> M offset wm*32
  const int wn   = wave & 1;          // 0..1 -> N offset wn*64
  const int bm   = blockIdx.y;
  const int bn   = blockIdx.x;

  // K-pair byte layout of 16-bit A/B fragments (ISA 7.12.2)
  int kofs[8];
#pragma unroll
  for (int v = 0; v < 8; ++v)
    kofs[v] = ((v < 4) ? 2 * v : 16 + 2 * (v - 4)) + 8 * half;

  // accumulators pre-loaded with bias (saves epilogue add + zero init)
  v8f acc[2][4];
#pragma unroll
  for (int j = 0; j < 4; ++j) {
    float bv = bias[bn * BN + wn * 64 + j * 16 + l];
#pragma unroll
    for (int i = 0; i < 2; ++i)
#pragma unroll
      for (int r = 0; r < 8; ++r) acc[i][j][r] = bv;
  }

  // ---- async copy streams: 2 chunks of A + 2 chunks of B per thread ----
  const int c0 = tid,        c1 = tid + 256;       // chunk ids (0..511)
  const int r0 = c0 >> 2,    q0 = (c0 & 3) * 8;    // row, col-of-8-halves
  const int r1 = c1 >> 2,    q1 = (c1 & 3) * 8;

  const unsigned short* gA0 = A  + (size_t)(bm * BM + r0) * lda + q0;
  const unsigned short* gA1 = A  + (size_t)(bm * BM + r1) * lda + q1;
  const unsigned short* gB0 = Bt + (size_t)(bn * BN + r0) * ldb + q0;
  const unsigned short* gB1 = Bt + (size_t)(bn * BN + r1) * ldb + q1;

  uint32_t lA0[2], lA1[2], lB0[2], lB1[2];
#pragma unroll
  for (int b = 0; b < 2; ++b) {
    lA0[b] = (uint32_t)(uintptr_t)&As[b][r0 * LDS_STR + q0];
    lA1[b] = (uint32_t)(uintptr_t)&As[b][r1 * LDS_STR + q1];
    lB0[b] = (uint32_t)(uintptr_t)&Bs[b][r0 * LDS_STR + q0];
    lB1[b] = (uint32_t)(uintptr_t)&Bs[b][r1 * LDS_STR + q1];
  }

  auto issue = [&](int buf) {
    asm volatile("global_load_async_to_lds_b128 %0, %1, off"
                 :: "v"(lA0[buf]), "v"(gA0) : "memory");
    asm volatile("global_load_async_to_lds_b128 %0, %1, off"
                 :: "v"(lA1[buf]), "v"(gA1) : "memory");
    asm volatile("global_load_async_to_lds_b128 %0, %1, off"
                 :: "v"(lB0[buf]), "v"(gB0) : "memory");
    asm volatile("global_load_async_to_lds_b128 %0, %1, off"
                 :: "v"(lB1[buf]), "v"(gB1) : "memory");
    gA0 += BK; gA1 += BK; gB0 += BK; gB1 += BK;
  };

  auto compute_tile = [&](int buf) {
    __syncthreads();                           // tile `buf` fully in LDS
    v16bf afrag[2], bfrag[4];
#pragma unroll
    for (int im = 0; im < 2; ++im) {
      const unsigned short* base = &As[buf][(wm * 32 + im * 16 + l) * LDS_STR];
      v8u u;
#pragma unroll
      for (int v = 0; v < 8; ++v) u[v] = *(const uint32_t*)(base + kofs[v]);
      afrag[im] = __builtin_bit_cast(v16bf, u);
    }
#pragma unroll
    for (int in = 0; in < 4; ++in) {
      const unsigned short* base = &Bs[buf][(wn * 64 + in * 16 + l) * LDS_STR];
      v8u u;
#pragma unroll
      for (int v = 0; v < 8; ++v) u[v] = *(const uint32_t*)(base + kofs[v]);
      bfrag[in] = __builtin_bit_cast(v16bf, u);
    }
#pragma unroll
    for (int im = 0; im < 2; ++im)
#pragma unroll
      for (int in = 0; in < 4; ++in)
        acc[im][in] = __builtin_amdgcn_wmma_f32_16x16x32_bf16(
            false, afrag[im], false, bfrag[in], (short)0, acc[im][in], false, false);
    __syncthreads();                           // everyone done reading `buf`
  };

  const int nk = K / BK;                       // even by construction
  issue(0);                                    // tile 0 -> buf0
  int k = 0;
  for (; k + 2 < nk; k += 2) {
    issue(1);                                  // tile k+1 -> buf1
    asm volatile("s_wait_asynccnt 4" ::: "memory");
    compute_tile(0);
    issue(0);                                  // tile k+2 -> buf0
    asm volatile("s_wait_asynccnt 4" ::: "memory");
    compute_tile(1);
  }
  issue(1);                                    // tile nk-1 -> buf1
  asm volatile("s_wait_asynccnt 4" ::: "memory");
  compute_tile(0);
  asm volatile("s_wait_asynccnt 0" ::: "memory");
  compute_tile(1);

  // ---- epilogue: relu + store (C layout: m = r + 8*half, n = l) ----
  const int mBase = bm * BM + wm * 32;
  const int nBase = bn * BN + wn * 64;
  if (bm * BM + BM <= Mstore) {           // uniform fast path: no guards
#pragma unroll
    for (int in = 0; in < 4; ++in) {
      int n = nBase + in * 16 + l;
#pragma unroll
      for (int im = 0; im < 2; ++im)
#pragma unroll
        for (int r = 0; r < 8; ++r) {
          int m = mBase + im * 16 + r + 8 * half;
          float v = acc[im][in][r];
          if (RELU) v = v > 0.0f ? v : 0.0f;
          if (OUT_BF16) ((unsigned short*)Cout)[(size_t)m * ldc + n] = f2bf(v);
          else          ((float*)Cout)[(size_t)m * ldc + n] = v;
        }
    }
  } else {
#pragma unroll
    for (int in = 0; in < 4; ++in) {
      int n = nBase + in * 16 + l;
#pragma unroll
      for (int im = 0; im < 2; ++im)
#pragma unroll
        for (int r = 0; r < 8; ++r) {
          int m = mBase + im * 16 + r + 8 * half;
          if (m < Mstore) {
            float v = acc[im][in][r];
            if (RELU) v = v > 0.0f ? v : 0.0f;
            if (OUT_BF16) ((unsigned short*)Cout)[(size_t)m * ldc + n] = f2bf(v);
            else          ((float*)Cout)[(size_t)m * ldc + n] = v;
          }
        }
    }
  }
}

// ---------------- elementwise / glue kernels ----------------
// weight cast WITH transpose: out[n*Kpad + k] = bf16(W[k*N + n]), zero K-pad
__global__ void cast_weight_T_kernel(const float* __restrict__ W,
                                     unsigned short* __restrict__ out,
                                     int K, int Kpad, int N) {
  size_t i = (size_t)blockIdx.x * blockDim.x + threadIdx.x;
  size_t total = (size_t)N * Kpad;
  if (i >= total) return;
  int n = (int)(i / Kpad), k = (int)(i % Kpad);
  out[i] = f2bf(k < K ? W[(size_t)k * N + n] : 0.0f);
}

__global__ void concat_embed_kernel(const float* __restrict__ emb, int embW,
                                    const float* __restrict__ table, int tabW,
                                    const int* __restrict__ idx, int idxStride, int idxOff,
                                    unsigned short* __restrict__ out, int outW,
                                    int M, int Mpad) {
  size_t i = (size_t)blockIdx.x * blockDim.x + threadIdx.x;
  size_t total = (size_t)Mpad * outW;
  if (i >= total) return;
  int row = (int)(i / outW), col = (int)(i % outW);
  float v = 0.0f;
  if (row < M) {
    if (col < embW) v = emb[(size_t)row * embW + col];
    else if (col < embW + tabW)
      v = table[(size_t)idx[row * idxStride + idxOff] * tabW + (col - embW)];
  }
  out[i] = f2bf(v);
}

__global__ void concat3_kernel(const unsigned short* __restrict__ objv, int objStride,
                               const unsigned short* __restrict__ predv, int predStride,
                               const int* __restrict__ rels, int W,
                               unsigned short* __restrict__ out, int T, int Tpad) {
  size_t i = (size_t)blockIdx.x * blockDim.x + threadIdx.x;
  size_t total = (size_t)Tpad * 3 * W;
  if (i >= total) return;
  int row = (int)(i / (3 * W)), col = (int)(i % (3 * W));
  unsigned short v = 0;
  if (row < T) {
    if (col < W)            v = objv[(size_t)rels[row * 3 + 0] * objStride + col];
    else if (col < 2 * W)   v = predv[(size_t)row * predStride + (col - W)];
    else                    v = objv[(size_t)rels[row * 3 + 2] * objStride + (col - 2 * W)];
  }
  out[i] = v;
}

__global__ void zero_f32_kernel(float* __restrict__ p, size_t n) {
  size_t i = (size_t)blockIdx.x * blockDim.x + threadIdx.x;
  if (i < n) p[i] = 0.0f;
}

__global__ void count_kernel(const int* __restrict__ rels, float* __restrict__ cnt, int T) {
  int i = blockIdx.x * blockDim.x + threadIdx.x;
  if (i < T) {
    atomicAdd(&cnt[rels[i * 3 + 0]], 1.0f);
    atomicAdd(&cnt[rels[i * 3 + 2]], 1.0f);
  }
}

__global__ void scatter_kernel(const float* __restrict__ t, const int* __restrict__ rels,
                               float* __restrict__ pooled, int T) {
  size_t i = (size_t)blockIdx.x * blockDim.x + threadIdx.x;
  if (i >= (size_t)T * DG) return;
  int r = (int)(i >> 9), c = (int)(i & (DG - 1));
  atomicAdd(&pooled[(size_t)rels[r * 3 + 0] * DG + c], t[(size_t)r * (3 * DG) + c]);
  atomicAdd(&pooled[(size_t)rels[r * 3 + 2] * DG + c], t[(size_t)r * (3 * DG) + 2 * DG + c]);
}

__global__ void normalize_cast_kernel(const float* __restrict__ pooled,
                                      const float* __restrict__ cnt,
                                      unsigned short* __restrict__ out, int M, int Mpad) {
  size_t i = (size_t)blockIdx.x * blockDim.x + threadIdx.x;
  if (i >= (size_t)Mpad * DG) return;
  int r = (int)(i >> 9);
  float v = 0.0f;
  if (r < M) {
    float c = cnt[r];
    v = pooled[i] / (c > 1.0f ? c : 1.0f);
  }
  out[i] = f2bf(v);
}

__global__ void slice_cast_kernel(const float* __restrict__ t,
                                  unsigned short* __restrict__ out, int T, int Tpad) {
  size_t i = (size_t)blockIdx.x * blockDim.x + threadIdx.x;
  if (i >= (size_t)Tpad * DG) return;
  int r = (int)(i >> 9), c = (int)(i & (DG - 1));
  float v = (r < T) ? t[(size_t)r * (3 * DG) + DG + c] : 0.0f;
  out[i] = f2bf(v);
}

__global__ void copy_pred_kernel(const float* __restrict__ t, float* __restrict__ out, int T) {
  size_t i = (size_t)blockIdx.x * blockDim.x + threadIdx.x;
  if (i >= (size_t)T * DG) return;
  int r = (int)(i >> 9), c = (int)(i & (DG - 1));
  out[i] = t[(size_t)r * (3 * DG) + DG + c];
}

// ---------------- host orchestration ----------------
static inline size_t cdiv(size_t a, size_t b) { return (a + b - 1) / b; }

extern "C" void kernel_launch(void* const* d_in, const int* in_sizes, int n_in,
                              void* d_out, int out_size, void* d_ws, size_t ws_size,
                              hipStream_t stream) {
  (void)in_sizes; (void)n_in; (void)out_size; (void)ws_size;

  const float* obj_embs  = (const float*)d_in[0];
  const float* pred_embs = (const float*)d_in[2];
  const int*   rels      = (const int*)d_in[4];
  const int*   objs      = (const int*)d_in[5];
  const float* obj_table = (const float*)d_in[6];
  const float* rel_table = (const float*)d_in[7];
  const float* Wf_obj = (const float*)d_in[8];   const float* bf_obj = (const float*)d_in[9];
  const float* Wf_rel = (const float*)d_in[10];  const float* bf_rel = (const float*)d_in[11];
  const float* W1a = (const float*)d_in[12];     const float* b1a = (const float*)d_in[13];
  const float* W1b = (const float*)d_in[14];     const float* b1b = (const float*)d_in[15];
  const float* W2a = (const float*)d_in[16];     const float* b2a = (const float*)d_in[17];
  const float* W2b = (const float*)d_in[18];     const float* b2b = (const float*)d_in[19];
  const float* Ws1a = (const float*)d_in[20];    const float* bs1a = (const float*)d_in[21];
  const float* Ws1b = (const float*)d_in[22];    const float* bs1b = (const float*)d_in[23];
  const float* Ws2a = (const float*)d_in[24];    const float* bs2a = (const float*)d_in[25];
  const float* Ws2b = (const float*)d_in[26];    const float* bs2b = (const float*)d_in[27];

  char* ws = (char*)d_ws;
  size_t off = 0;
  auto alloc = [&](size_t bytes) -> void* {
    void* p = ws + off;
    off = (off + bytes + 255) & ~(size_t)255;
    return p;
  };

  // bf16 transposed weight buffers (N x Kpad)
  unsigned short* Wobj_bf = (unsigned short*)alloc((size_t)DIN * KFP * 2);
  unsigned short* Wrel_bf = (unsigned short*)alloc((size_t)DIN * KFP * 2);
  unsigned short* W1a_bf  = (unsigned short*)alloc((size_t)HH * 3 * DIN * 2);
  unsigned short* W1b_bf  = (unsigned short*)alloc((size_t)3 * DG * HH * 2);
  unsigned short* W2a_bf  = (unsigned short*)alloc((size_t)HH * HH * 2);
  unsigned short* W2b_bf  = (unsigned short*)alloc((size_t)DG * HH * 2);
  unsigned short* Ws1a_bf = (unsigned short*)alloc((size_t)L_EXTRA * HH * 3 * DG * 2);
  unsigned short* Ws1b_bf = (unsigned short*)alloc((size_t)L_EXTRA * 3 * DG * HH * 2);
  unsigned short* Ws2a_bf = (unsigned short*)alloc((size_t)L_EXTRA * HH * HH * 2);
  unsigned short* Ws2b_bf = (unsigned short*)alloc((size_t)L_EXTRA * DG * HH * 2);

  // bf16 activations
  unsigned short* Xobj      = (unsigned short*)alloc((size_t)OPAD * KFP * 2);
  unsigned short* Xpred     = (unsigned short*)alloc((size_t)TPAD * KFP * 2);
  unsigned short* objv_big  = (unsigned short*)alloc((size_t)OPAD * DIN * 2);
  unsigned short* predv_big = (unsigned short*)alloc((size_t)TPAD * DIN * 2);
  unsigned short* cur       = (unsigned short*)alloc((size_t)TPAD * 3 * DIN * 2);
  unsigned short* hbuf      = (unsigned short*)alloc((size_t)TPAD * HH * 2);
  unsigned short* objv      = (unsigned short*)alloc((size_t)OPAD * DG * 2);
  unsigned short* h2buf     = (unsigned short*)alloc((size_t)OPAD * HH * 2);
  unsigned short* pooled_bf = (unsigned short*)alloc((size_t)OPAD * DG * 2);
  unsigned short* predv     = (unsigned short*)alloc((size_t)TPAD * DG * 2);
  // f32 scratch
  float* t_f32  = (float*)alloc((size_t)TPAD * 3 * DG * 4);
  float* pooled = (float*)alloc((size_t)OPAD * DG * 4);
  float* cnt    = (float*)alloc((size_t)OPAD * 4);

  const int EW = 256;
  auto ew = [&](size_t total) { return dim3((unsigned)cdiv(total, EW)); };

  auto castWT = [&](const float* src, unsigned short* dst, int K, int Kpad, int N) {
    cast_weight_T_kernel<<<ew((size_t)N * Kpad), EW, 0, stream>>>(src, dst, K, Kpad, N);
  };
  auto gemm_bf = [&](const unsigned short* A, int lda, const unsigned short* Bt, int ldb,
                     const float* bias, unsigned short* C, int ldc,
                     int Mpad, int N, int K) {
    dim3 grid(N / BN, Mpad / BM);
    gemm_bf16_wmma<true, true><<<grid, 256, 0, stream>>>(A, lda, Bt, ldb, bias, C, ldc, Mpad, K);
  };
  auto gemm_f32 = [&](const unsigned short* A, int lda, const unsigned short* Bt, int ldb,
                      const float* bias, float* C, int ldc,
                      int Mpad, int Mstore, int N, int K) {
    dim3 grid(N / BN, Mpad / BM);
    gemm_bf16_wmma<true, false><<<grid, 256, 0, stream>>>(A, lda, Bt, ldb, bias, C, ldc, Mstore, K);
  };

  // ---- weight conversion (transposed) ----
  castWT(Wf_obj, Wobj_bf, KF, KFP, DIN);
  castWT(Wf_rel, Wrel_bf, KF, KFP, DIN);
  castWT(W1a, W1a_bf, 3 * DIN, 3 * DIN, HH);
  castWT(W1b, W1b_bf, HH, HH, 3 * DG);
  castWT(W2a, W2a_bf, HH, HH, HH);
  castWT(W2b, W2b_bf, HH, HH, DG);
  for (int i = 0; i < L_EXTRA; ++i) {
    castWT(Ws1a + (size_t)i * 3 * DG * HH, Ws1a_bf + (size_t)i * HH * 3 * DG, 3 * DG, 3 * DG, HH);
    castWT(Ws1b + (size_t)i * HH * 3 * DG, Ws1b_bf + (size_t)i * 3 * DG * HH, HH, HH, 3 * DG);
    castWT(Ws2a + (size_t)i * HH * HH,     Ws2a_bf + (size_t)i * HH * HH,     HH, HH, HH);
    castWT(Ws2b + (size_t)i * HH * DG,     Ws2b_bf + (size_t)i * DG * HH,     HH, HH, DG);
  }

  // ---- input embeddings (concat + bf16 + pad) ----
  concat_embed_kernel<<<ew((size_t)OPAD * KFP), EW, 0, stream>>>(
      obj_embs, DIN, obj_table, DW, objs, 1, 0, Xobj, KFP, O_N, OPAD);
  concat_embed_kernel<<<ew((size_t)TPAD * KFP), EW, 0, stream>>>(
      pred_embs, DIN, rel_table, DW, rels, 3, 1, Xpred, KFP, T_N, TPAD);

  // ---- layer 0 projections ----
  gemm_bf(Xobj, KFP, Wobj_bf, KFP, bf_obj, objv_big, DIN, OPAD, DIN, KFP);
  gemm_bf(Xpred, KFP, Wrel_bf, KFP, bf_rel, predv_big, DIN, TPAD, DIN, KFP);

  // ---- degree counts (same every layer) ----
  zero_f32_kernel<<<ew(OPAD), EW, 0, stream>>>(cnt, OPAD);
  count_kernel<<<ew(T_N), EW, 0, stream>>>(rels, cnt, T_N);

  // ---- triple-conv layer 1 (wide, 2048-dim obj/pred) ----
  concat3_kernel<<<ew((size_t)TPAD * 3 * DIN), EW, 0, stream>>>(
      objv_big, DIN, predv_big, DIN, rels, DIN, cur, T_N, TPAD);
  gemm_bf(cur, 3 * DIN, W1a_bf, 3 * DIN, b1a, hbuf, HH, TPAD, HH, 3 * DIN);
  gemm_f32(hbuf, HH, W1b_bf, HH, b1b, t_f32, 3 * DG, TPAD, TPAD, 3 * DG, HH);
  zero_f32_kernel<<<ew((size_t)OPAD * DG), EW, 0, stream>>>(pooled, (size_t)OPAD * DG);
  scatter_kernel<<<ew((size_t)T_N * DG), EW, 0, stream>>>(t_f32, rels, pooled, T_N);
  normalize_cast_kernel<<<ew((size_t)OPAD * DG), EW, 0, stream>>>(pooled, cnt, pooled_bf, O_N, OPAD);
  slice_cast_kernel<<<ew((size_t)TPAD * DG), EW, 0, stream>>>(t_f32, predv, T_N, TPAD);
  gemm_bf(pooled_bf, DG, W2a_bf, DG, b2a, h2buf, HH, OPAD, HH, DG);
  gemm_bf(h2buf, HH, W2b_bf, HH, b2b, objv, DG, OPAD, DG, HH);

  // ---- extra layers ----
  for (int i = 0; i < L_EXTRA; ++i) {
    bool last = (i == L_EXTRA - 1);
    const unsigned short* w1a = Ws1a_bf + (size_t)i * HH * 3 * DG;
    const unsigned short* w1b = Ws1b_bf + (size_t)i * 3 * DG * HH;
    const unsigned short* w2a = Ws2a_bf + (size_t)i * HH * HH;
    const unsigned short* w2b = Ws2b_bf + (size_t)i * DG * HH;
    const float* bb1a = bs1a + (size_t)i * HH;
    const float* bb1b = bs1b + (size_t)i * 3 * DG;
    const float* bb2a = bs2a + (size_t)i * HH;
    const float* bb2b = bs2b + (size_t)i * DG;

    concat3_kernel<<<ew((size_t)TPAD * 3 * DG), EW, 0, stream>>>(
        objv, DG, predv, DG, rels, DG, cur, T_N, TPAD);
    gemm_bf(cur, 3 * DG, w1a, 3 * DG, bb1a, hbuf, HH, TPAD, HH, 3 * DG);
    gemm_f32(hbuf, HH, w1b, HH, bb1b, t_f32, 3 * DG, TPAD, TPAD, 3 * DG, HH);
    zero_f32_kernel<<<ew((size_t)OPAD * DG), EW, 0, stream>>>(pooled, (size_t)OPAD * DG);
    scatter_kernel<<<ew((size_t)T_N * DG), EW, 0, stream>>>(t_f32, rels, pooled, T_N);
    normalize_cast_kernel<<<ew((size_t)OPAD * DG), EW, 0, stream>>>(pooled, cnt, pooled_bf, O_N, OPAD);
    if (!last) {
      slice_cast_kernel<<<ew((size_t)TPAD * DG), EW, 0, stream>>>(t_f32, predv, T_N, TPAD);
    } else {
      copy_pred_kernel<<<ew((size_t)T_N * DG), EW, 0, stream>>>(
          t_f32, (float*)d_out + (size_t)O_N * DG, T_N);
    }
    gemm_bf(pooled_bf, DG, w2a, DG, bb2a, h2buf, HH, OPAD, HH, DG);
    if (!last) {
      gemm_bf(h2buf, HH, w2b, HH, bb2b, objv, DG, OPAD, DG, HH);
    } else {
      gemm_f32(h2buf, HH, w2b, HH, bb2b, (float*)d_out, DG, OPAD, O_N, DG, HH);
    }
  }
}
